// Kelso_Model_43121471652517
// MI455X (gfx1250) — matrix operand order repaired and checked by reading, not compiled, not run on hardware
//
#include <hip/hip_runtime.h>
#include <hip/hip_bf16.h>

static constexpr int B  = 2;
static constexpr int S  = 2048;
static constexpr int H  = 1024;
static constexpr int NH = 16;
static constexpr int HD = 64;

typedef unsigned short u16;
typedef __attribute__((ext_vector_type(8)))  u16    u16x8;
typedef __attribute__((ext_vector_type(16))) u16    u16x16;
typedef __attribute__((ext_vector_type(16))) __bf16 bf16x16;
typedef __attribute__((ext_vector_type(8)))  float  f32x8;
typedef __attribute__((ext_vector_type(4)))  unsigned int u32x4;
typedef __attribute__((ext_vector_type(4)))  int    i32x4;
typedef __attribute__((ext_vector_type(8)))  int    i32x8;

#if __has_builtin(__builtin_amdgcn_tensor_load_to_lds)
#define USE_TDM 1
#else
#define USE_TDM 0
#endif

#define NEGBIG (-3.0e38f)

__device__ __forceinline__ u16 f2bf(float f) {
    unsigned u = __float_as_uint(f);
    unsigned r = u + 0x7fffu + ((u >> 16) & 1u);   // round-to-nearest-even
    return (u16)(r >> 16);
}

__device__ __forceinline__ u16x16 cat8(u16x8 a, u16x8 b) {
    return __builtin_shufflevector(a, b, 0,1,2,3,4,5,6,7,8,9,10,11,12,13,14,15);
}

__device__ __forceinline__ f32x8 bwmma(u16x16 a, u16x16 b, f32x8 c) {
    return __builtin_amdgcn_wmma_f32_16x16x32_bf16(
        false, __builtin_bit_cast(bf16x16, a),
        false, __builtin_bit_cast(bf16x16, b),
        (short)0, c, false, false);
}

// --- TDM: DMA a 2D tile (tile_x elems wide, tile_y rows, 2-byte elems) ------
// Builds the D# (groups 0/1 per CDNA5 ISA 8.3/8.4) and issues tensor_load_to_lds.
__device__ __forceinline__ void tdm_load_2d(unsigned lds_byte_off,
                                            const void* gptr,
                                            unsigned tile_x, unsigned tile_y,
                                            unsigned stride_elems) {
#if USE_TDM
    unsigned long long ga = (unsigned long long)(size_t)gptr;
    u32x4 g0;
    g0[0] = 1u;                                        // count=1 (valid), user mode
    g0[1] = lds_byte_off;                              // lds_addr
    g0[2] = (unsigned)ga;                              // global_addr[31:0]
    g0[3] = (unsigned)((ga >> 32) & 0x01ffffffu)       // global_addr[56:32]
          | (2u << 30);                                // type=2 ("image")
    i32x8 g1;
    g1[0] = (int)(1u << 16);                           // data_size=1 -> 2 bytes
    g1[1] = (int)((tile_x & 0xffffu) << 16);           // tensor_dim0[15:0]
    g1[2] = (int)(((tile_x >> 16) & 0xffffu)           // tensor_dim0[31:16]
          | ((tile_y & 0xffffu) << 16));               // tensor_dim1[15:0]
    g1[3] = (int)(((tile_y >> 16) & 0xffffu)           // tensor_dim1[31:16]
          | ((tile_x & 0xffffu) << 16));               // tile_dim0
    g1[4] = (int)(tile_y & 0xffffu);                   // tile_dim1 (tile_dim2=0)
    g1[5] = (int)stride_elems;                         // tensor_dim0_stride[31:0]
    g1[6] = 0;
    g1[7] = 0;
    i32x4 z4; z4[0] = 0; z4[1] = 0; z4[2] = 0; z4[3] = 0;
#if __clang_major__ >= 23
    i32x8 z8; z8[0]=0; z8[1]=0; z8[2]=0; z8[3]=0; z8[4]=0; z8[5]=0; z8[6]=0; z8[7]=0;
    __builtin_amdgcn_tensor_load_to_lds(g0, g1, z4, z4, z8, 0);
#else
    __builtin_amdgcn_tensor_load_to_lds(g0, g1, z4, z4, 0);
#endif
#else
    (void)lds_byte_off; (void)gptr; (void)tile_x; (void)tile_y; (void)stride_elems;
#endif
}

// ---------------- weight fp32 -> bf16, transposed to [N][K] -----------------
__global__ __launch_bounds__(256) void wtrans_kernel(const float* __restrict__ W,
                                                     u16* __restrict__ Wt) {
    int idx = blockIdx.x * 256 + threadIdx.x;      // over 1024*1024, idx = n*1024+k
    int n = idx >> 10, k = idx & 1023;
    Wt[idx] = f2bf(W[(size_t)k * 1024 + n]);
}

// ---------------- embedding gather -> bf16 ----------------------------------
__global__ __launch_bounds__(256) void embed_kernel(const int* __restrict__ batch,
                                                    const float* __restrict__ emb,
                                                    u16* __restrict__ hbf) {
    int idx = blockIdx.x * 256 + threadIdx.x;      // group of 4 elements
    int col = (idx & (H / 4 - 1)) * 4;
    int tok = idx >> 8;                            // H/4 == 256
    int row = batch[tok];
    float4 v = *(const float4*)&emb[(size_t)row * H + col];
    u16* o = &hbf[(size_t)tok * H + col];
    o[0] = f2bf(v.x); o[1] = f2bf(v.y); o[2] = f2bf(v.z); o[3] = f2bf(v.w);
}

// ---------------- tiled bf16 GEMM: C[M,N] = A[M,K] * Bt[N,K]^T + bias -------
// TDM path: double-buffered LDS tiles, wave 0 issues async tensor DMA while
// all 8 waves run WMMA on the live buffer; one barrier per K-step.
__global__ __launch_bounds__(256) void gemm_kernel(const u16* __restrict__ A,
                                                   const u16* __restrict__ Bt,
                                                   const float* __restrict__ bias,
                                                   float* __restrict__ C,
                                                   int M, int N, int K) {
#if USE_TDM
    __shared__ __attribute__((aligned(16))) u16 As[2][128 * 32];
    __shared__ __attribute__((aligned(16))) u16 Bs[2][128 * 32];
#else
    __shared__ __attribute__((aligned(16))) u16 As[1][128 * 32];
    __shared__ __attribute__((aligned(16))) u16 Bs[1][128 * 32];
#endif
    const int tid  = threadIdx.x;
    const int lane = tid & 31, wave = tid >> 5;
    const int hl   = lane >> 4, lr = lane & 15;
    const int bm = blockIdx.y * 128, bn = blockIdx.x * 128;
    const int wm = (wave & 1) * 64, wn = (wave >> 1) * 32;
    (void)M;

    f32x8 acc[4][2];
    #pragma unroll
    for (int i = 0; i < 4; i++)
        #pragma unroll
        for (int j = 0; j < 2; j++)
            #pragma unroll
            for (int r = 0; r < 8; r++) acc[i][j][r] = 0.0f;

    int cur = 0;
#if USE_TDM
    if (wave == 0) {
        tdm_load_2d((unsigned)(size_t)(void*)&As[0][0], A + (size_t)bm * K, 32, 128, K);
        tdm_load_2d((unsigned)(size_t)(void*)&Bs[0][0], Bt + (size_t)bn * K, 32, 128, K);
    }
#endif

    for (int k0 = 0; k0 < K; k0 += 32) {
#if USE_TDM
        if (wave == 0) __builtin_amdgcn_s_wait_tensorcnt(0);
        __syncthreads();   // publish buffer `cur`; retire reads of buffer `cur^1`
        if (wave == 0 && (k0 + 32) < K) {
            tdm_load_2d((unsigned)(size_t)(void*)&As[cur ^ 1][0],
                        A + (size_t)bm * K + (k0 + 32), 32, 128, K);
            tdm_load_2d((unsigned)(size_t)(void*)&Bs[cur ^ 1][0],
                        Bt + (size_t)bn * K + (k0 + 32), 32, 128, K);
        }
#else
        #pragma unroll
        for (int i = 0; i < 2; i++) {
            int c = tid + i * 256;              // 0..511
            int row = c >> 2;                   // 0..127
            int col = (c & 3) * 8;              // 0,8,16,24
            *(u16x8*)&As[0][row * 32 + col] = *(const u16x8*)&A[(size_t)(bm + row) * K + k0 + col];
            *(u16x8*)&Bs[0][row * 32 + col] = *(const u16x8*)&Bt[(size_t)(bn + row) * K + k0 + col];
            if (k0 + 32 < K) {
                __builtin_prefetch(&A[(size_t)(bm + row) * K + k0 + 32 + col], 0, 0);
                __builtin_prefetch(&Bt[(size_t)(bn + row) * K + k0 + 32 + col], 0, 0);
            }
        }
        __syncthreads();
#endif

        u16x16 bfrag[2];
        #pragma unroll
        for (int n16 = 0; n16 < 2; n16++) {
            const u16* p = &Bs[cur][(wn + n16 * 16 + lr) * 32 + hl * 16];
            bfrag[n16] = cat8(*(const u16x8*)p, *(const u16x8*)(p + 8));
        }
        #pragma unroll
        for (int m16 = 0; m16 < 4; m16++) {
            const u16* p = &As[cur][(wm + m16 * 16 + lr) * 32 + hl * 16];
            u16x16 af = cat8(*(const u16x8*)p, *(const u16x8*)(p + 8));
            #pragma unroll
            for (int n16 = 0; n16 < 2; n16++)
                acc[m16][n16] = bwmma(af, bfrag[n16], acc[m16][n16]);
        }
#if USE_TDM
        cur ^= 1;
#else
        __syncthreads();
#endif
    }

    #pragma unroll
    for (int n16 = 0; n16 < 2; n16++) {
        int n = bn + wn + n16 * 16 + lr;
        float bb = bias ? bias[n] : 0.0f;
        #pragma unroll
        for (int m16 = 0; m16 < 4; m16++) {
            #pragma unroll
            for (int r = 0; r < 8; r++) {
                int m = bm + wm + m16 * 16 + hl * 8 + r;
                C[(size_t)m * N + n] = acc[m16][n16][r] + bb;
            }
        }
    }
}

// ---------------- RoPE + layout: q,k -> [B,NH,S,HD] bf16 ; v -> [B,NH,HD,S] -
__global__ __launch_bounds__(256) void rope_kernel(const float* __restrict__ qf,
                                                   const float* __restrict__ kf,
                                                   const float* __restrict__ vf,
                                                   u16* __restrict__ qh,
                                                   u16* __restrict__ kh,
                                                   u16* __restrict__ vt) {
    int idx = blockIdx.x * 256 + threadIdx.x;      // over B*S*NH*32
    int d = idx & 31;
    int t = idx >> 5;
    int h = t % NH;
    int s = (t / NH) % S;
    int b = t / (NH * S);
    size_t src = ((size_t)b * S + s) * (NH * HD) + h * HD + d;
    float inv = __powf(10000.0f, -(float)d / 32.0f);
    float ang = (float)s * inv;
    float sn, cs;
    __sincosf(ang, &sn, &cs);

    float q1 = qf[src], q2 = qf[src + 32];
    float k1 = kf[src], k2 = kf[src + 32];
    const float sc = 0.125f;                       // 1/sqrt(HD) folded into q
    size_t dst = (((size_t)b * NH + h) * S + s) * HD + d;
    qh[dst]      = f2bf((q1 * cs - q2 * sn) * sc);
    qh[dst + 32] = f2bf((q2 * cs + q1 * sn) * sc);
    kh[dst]      = f2bf(k1 * cs - k2 * sn);
    kh[dst + 32] = f2bf(k2 * cs + k1 * sn);

    size_t vd = (((size_t)b * NH + h) * HD + d) * S + s;
    vt[vd]                  = f2bf(vf[src]);
    vt[vd + 32 * (size_t)S] = f2bf(vf[src + 32]);
}

// ---------------- flash attention (causal + pad), bf16 WMMA -----------------
__global__ __launch_bounds__(256) void attn_kernel(const u16* __restrict__ qh,
                                                   const u16* __restrict__ kh,
                                                   const u16* __restrict__ vt,
                                                   const int* __restrict__ lengths,
                                                   u16* __restrict__ out) {
    __shared__ __attribute__((aligned(16))) u16 Ps[8][16 * 32];  // per-wave P staging
    const int tid  = threadIdx.x;
    const int lane = tid & 31, wave = tid >> 5;
    const int hl   = lane >> 4, lr = lane & 15;
    const int b = blockIdx.z, h = blockIdx.y;
    const int q0 = blockIdx.x * 128 + wave * 16;
    const int len = lengths[b];

    const u16* Q  = qh + ((size_t)b * NH + h) * S * HD;
    const u16* Kp = kh + ((size_t)b * NH + h) * S * HD;
    const u16* Vt = vt + ((size_t)b * NH + h) * HD * S;

    // Q A-fragments for d=[0,32) and d=[32,64)
    u16x16 qa[2];
    #pragma unroll
    for (int kk = 0; kk < 2; kk++) {
        const u16* p = Q + (size_t)(q0 + lr) * HD + kk * 32 + hl * 16;
        qa[kk] = cat8(*(const u16x8*)p, *(const u16x8*)(p + 8));
    }

    float Mr[8], Lr[8];
    f32x8 oacc[4];
    #pragma unroll
    for (int r = 0; r < 8; r++) { Mr[r] = NEGBIG; Lr[r] = 0.0f; }
    #pragma unroll
    for (int dt = 0; dt < 4; dt++)
        #pragma unroll
        for (int r = 0; r < 8; r++) oacc[dt][r] = 0.0f;

    const int kend = q0 + 15;                       // causal bound for this wave
    for (int k0 = 0; k0 <= kend; k0 += 32) {
        // scores 16x32: two 16-col tiles
        f32x8 s0, s1;
        #pragma unroll
        for (int r = 0; r < 8; r++) { s0[r] = 0.0f; s1[r] = 0.0f; }
        #pragma unroll
        for (int kk = 0; kk < 2; kk++) {
            const u16* p0 = Kp + (size_t)(k0 + lr) * HD + kk * 32 + hl * 16;
            const u16* p1 = Kp + (size_t)(k0 + 16 + lr) * HD + kk * 32 + hl * 16;
            u16x16 kb0 = cat8(*(const u16x8*)p0, *(const u16x8*)(p0 + 8));
            u16x16 kb1 = cat8(*(const u16x8*)p1, *(const u16x8*)(p1 + 8));
            s0 = bwmma(qa[kk], kb0, s0);
            s1 = bwmma(qa[kk], kb1, s1);
        }

        // online softmax update per row
        const int c0 = k0 + lr, c1 = k0 + 16 + lr;
        #pragma unroll
        for (int r = 0; r < 8; r++) {
            int qpos = q0 + hl * 8 + r;
            float a0 = (c0 <= qpos && c0 < len) ? s0[r] : NEGBIG;
            float a1 = (c1 <= qpos && c1 < len) ? s1[r] : NEGBIG;
            float cm = fmaxf(a0, a1);
            #pragma unroll
            for (int m = 1; m < 16; m <<= 1) cm = fmaxf(cm, __shfl_xor(cm, m, 32));
            float newM  = fmaxf(Mr[r], cm);
            float alpha = __expf(Mr[r] - newM);
            float p0 = __expf(a0 - newM);
            float p1 = __expf(a1 - newM);
            float ps = p0 + p1;
            #pragma unroll
            for (int m = 1; m < 16; m <<= 1) ps += __shfl_xor(ps, m, 32);
            Lr[r] = Lr[r] * alpha + ps;
            Mr[r] = newM;
            #pragma unroll
            for (int dt = 0; dt < 4; dt++) oacc[dt][r] *= alpha;
            int mrow = hl * 8 + r;
            Ps[wave][mrow * 32 + lr]      = f2bf(p0);
            Ps[wave][mrow * 32 + 16 + lr] = f2bf(p1);
        }

        // P (A-frag from LDS) x V (B-frag from transposed global V)
        const u16* pp = &Ps[wave][lr * 32 + hl * 16];
        u16x16 pa = cat8(*(const u16x8*)pp, *(const u16x8*)(pp + 8));
        #pragma unroll
        for (int dt = 0; dt < 4; dt++) {
            const u16* pv = Vt + (size_t)(dt * 16 + lr) * S + k0 + hl * 16;
            u16x16 vb = cat8(*(const u16x8*)pv, *(const u16x8*)(pv + 8));
            oacc[dt] = bwmma(pa, vb, oacc[dt]);
        }
    }

    // normalize + store bf16 to [B,S,NH*HD]
    #pragma unroll
    for (int r = 0; r < 8; r++) {
        float inv = 1.0f / Lr[r];
        int qpos = q0 + hl * 8 + r;
        size_t base = ((size_t)b * S + qpos) * (NH * HD) + h * HD;
        #pragma unroll
        for (int dt = 0; dt < 4; dt++)
            out[base + dt * 16 + lr] = f2bf(oacc[dt][r] * inv);
    }
}

// ---------------------------------------------------------------------------
extern "C" void kernel_launch(void* const* d_in, const int* in_sizes, int n_in,
                              void* d_out, int out_size, void* d_ws, size_t ws_size,
                              hipStream_t stream) {
    (void)in_sizes; (void)n_in; (void)out_size; (void)ws_size;
    const int*   batch   = (const int*)  d_in[0];
    const int*   lengths = (const int*)  d_in[2];   // d_in[1] positions == arange(S)
    const float* emb     = (const float*)d_in[3];
    const float* wq = (const float*)d_in[4];
    const float* bq = (const float*)d_in[5];
    const float* wk = (const float*)d_in[6];
    const float* bk = (const float*)d_in[7];
    const float* wv = (const float*)d_in[8];
    const float* bv = (const float*)d_in[9];
    const float* wo = (const float*)d_in[10];
    const float* bo = (const float*)d_in[11];
    float* out = (float*)d_out;

    size_t off = 0;
    auto carve = [&](size_t bytes) -> void* {
        off = (off + 255) & ~(size_t)255;
        void* p = (char*)d_ws + off;
        off += bytes;
        return p;
    };
    const size_t BS = (size_t)B * S;                 // 4096
    u16*   hbf = (u16*)carve(BS * H * 2);
    u16*   wqt = (u16*)carve((size_t)H * H * 2);
    u16*   wkt = (u16*)carve((size_t)H * H * 2);
    u16*   wvt = (u16*)carve((size_t)H * H * 2);
    u16*   wot = (u16*)carve((size_t)H * H * 2);
    float* qf  = (float*)carve(BS * H * 4);
    float* kf  = (float*)carve(BS * H * 4);
    float* vf  = (float*)carve(BS * H * 4);
    u16*   qhb = (u16*)carve(BS * H * 2);
    u16*   khb = (u16*)carve(BS * H * 2);
    u16*   vtb = (u16*)carve(BS * H * 2);
    u16*   aob = (u16*)carve(BS * H * 2);

    wtrans_kernel<<<(H * H) / 256, 256, 0, stream>>>(wq, wqt);
    wtrans_kernel<<<(H * H) / 256, 256, 0, stream>>>(wk, wkt);
    wtrans_kernel<<<(H * H) / 256, 256, 0, stream>>>(wv, wvt);
    wtrans_kernel<<<(H * H) / 256, 256, 0, stream>>>(wo, wot);

    embed_kernel<<<(int)(BS * H / 4 / 256), 256, 0, stream>>>(batch, emb, hbf);

    dim3 gg(H / 128, (int)(BS / 128));
    gemm_kernel<<<gg, 256, 0, stream>>>(hbf, wqt, bq, qf, (int)BS, H, H);
    gemm_kernel<<<gg, 256, 0, stream>>>(hbf, wkt, bk, kf, (int)BS, H, H);
    gemm_kernel<<<gg, 256, 0, stream>>>(hbf, wvt, bv, vf, (int)BS, H, H);

    rope_kernel<<<(int)(BS * NH * 32 / 256), 256, 0, stream>>>(qf, kf, vf, qhb, khb, vtb);

    attn_kernel<<<dim3(S / 128, NH, B), 256, 0, stream>>>(qhb, khb, vtb, lengths, aob);

    gemm_kernel<<<gg, 256, 0, stream>>>(aob, wot, bo, out, (int)BS, H, H);
}
